// Attention_867583394303
// MI455X (gfx1250) — matrix operand order repaired and checked by reading, not compiled, not run on hardware
//
#include <hip/hip_runtime.h>

#define DIM 768
#define NHEADS 12
#define HDIM 64
#define BB 4
#define NN 2048
#define ROWS (BB * NN)      /* 8192 */
#define QKV_DIM (3 * DIM)   /* 2304 */
#define SCALE_Q 0.125f      /* 64^-0.5 */

typedef unsigned int u32;
typedef unsigned short u16;
typedef __attribute__((ext_vector_type(4)))  u32    u32x4;
typedef __attribute__((ext_vector_type(8)))  u32    u32x8;
typedef __attribute__((ext_vector_type(16))) __bf16 v16bf;
typedef __attribute__((ext_vector_type(8)))  float  v8f;

// ---------- helpers ----------
static __device__ __forceinline__ u16 f2bf(float f) {
  u32 u = __builtin_bit_cast(u32, f);
  u32 r = u + 0x7FFFu + ((u >> 16) & 1u);   // round-to-nearest-even
  return (u16)(r >> 16);
}
static __device__ __forceinline__ u32 pack2(float lo, float hi) {
  return (u32)f2bf(lo) | ((u32)f2bf(hi) << 16);
}
// 16 bf16 operand: elements 0..7 at p[0..7], elements 8..15 at p[off2..off2+7]
static __device__ __forceinline__ v16bf ld16bf(const u16* p, int off2) {
  u32x4 lo = *(const u32x4*)p;
  u32x4 hi = *(const u32x4*)(p + off2);
  u32x8 w = __builtin_shufflevector(lo, hi, 0, 1, 2, 3, 4, 5, 6, 7);
  return __builtin_bit_cast(v16bf, w);
}
// xor-16 lane exchange within wave32 (ds_swizzle SWAPX16)
static __device__ __forceinline__ u32 swz16(u32 v) {
  return (u32)__builtin_amdgcn_ds_swizzle((int)v, 0x401F);
}
static __device__ __forceinline__ float swz16f(float v) {
  return __builtin_bit_cast(float, swz16(__builtin_bit_cast(u32, v)));
}
#define WMMA_BF16(A, B, C) \
  __builtin_amdgcn_wmma_f32_16x16x32_bf16(false, (A), false, (B), (short)0, (C), false, false)

// ---------- kernel 0: fp32 -> bf16 ----------
__global__ __launch_bounds__(256) void k_cvt_bf16(const float* __restrict__ src,
                                                  u16* __restrict__ dst, int n) {
  int i = blockIdx.x * 256 + threadIdx.x;
  if (i < n) dst[i] = f2bf(src[i]);
}

// ---------- kernel 1: QKV GEMM (32x64 tile/wave), scatter Q(scaled)/K/Vt ----------
__global__ __launch_bounds__(256) void k_qkv_gemm(const u16* __restrict__ xb,
                                                  const u16* __restrict__ wb,
                                                  u16* __restrict__ Qb,
                                                  u16* __restrict__ Kb,
                                                  u16* __restrict__ Vtb) {
  int wave = (blockIdx.x * 256 + threadIdx.x) >> 5;
  int lane = threadIdx.x & 31;
  int lrow = lane & 15, lhi = lane >> 4;
  int kaA = lhi * 16, kaB = lhi * 16;
  const int NSTRIP = QKV_DIM / 64;   // 36
  int m  = (wave / NSTRIP) * 32;     // 32-row x tile
  int n0 = (wave % NSTRIP) * 64;     // 64-col qkv strip

  v8f acc[2][4];
#pragma unroll
  for (int mi = 0; mi < 2; ++mi)
#pragma unroll
    for (int j = 0; j < 4; ++j)
#pragma unroll
      for (int r = 0; r < 8; ++r) acc[mi][j][r] = 0.f;

  const u16* arow0 = xb + (size_t)(m + lrow) * DIM + kaA;
  const u16* arow1 = arow0 + (size_t)16 * DIM;
  const u16* bbase = wb + (size_t)(n0 + lrow) * DIM + kaB;

  for (int kc = 0; kc < DIM; kc += 32) {
    // issue all operand loads up-front, then 8 WMMAs (partial loadcnt waits)
    v16bf A0 = ld16bf(arow0 + kc, 8);
    v16bf A1 = ld16bf(arow1 + kc, 8);
    v16bf B0 = ld16bf(bbase + kc, 8);
    v16bf B1 = ld16bf(bbase + (size_t)16 * DIM + kc, 8);
    v16bf B2 = ld16bf(bbase + (size_t)32 * DIM + kc, 8);
    v16bf B3 = ld16bf(bbase + (size_t)48 * DIM + kc, 8);
    acc[0][0] = WMMA_BF16(A0, B0, acc[0][0]);
    acc[1][0] = WMMA_BF16(A1, B0, acc[1][0]);
    acc[0][1] = WMMA_BF16(A0, B1, acc[0][1]);
    acc[1][1] = WMMA_BF16(A1, B1, acc[1][1]);
    acc[0][2] = WMMA_BF16(A0, B2, acc[0][2]);
    acc[1][2] = WMMA_BF16(A1, B2, acc[1][2]);
    acc[0][3] = WMMA_BF16(A0, B3, acc[0][3]);
    acc[1][3] = WMMA_BF16(A1, B3, acc[1][3]);
  }

#pragma unroll
  for (int mi = 0; mi < 2; ++mi) {
    int mbase = m + 16 * mi;
    int b = mbase / NN;
    int nloc = mbase - b * NN;
#pragma unroll
    for (int j = 0; j < 4; ++j) {
      int dbase = n0 + 16 * j;          // uniform per subtile
      int s  = dbase / DIM;             // 0=q, 1=k, 2=v
      int dd = dbase % DIM;
      int hh = dd / HDIM;
      int dh = (dd % HDIM) + lrow;
      size_t bh = (size_t)(b * NHEADS + hh);
      if (s == 0) {
        u16* q = Qb + bh * (size_t)NN * HDIM;
#pragma unroll
        for (int r = 0; r < 8; ++r)
          q[(size_t)(nloc + lhi * 8 + r) * HDIM + dh] = f2bf(acc[mi][j][r] * SCALE_Q);
      } else if (s == 1) {
        u16* kk = Kb + bh * (size_t)NN * HDIM;
#pragma unroll
        for (int r = 0; r < 8; ++r)
          kk[(size_t)(nloc + lhi * 8 + r) * HDIM + dh] = f2bf(acc[mi][j][r]);
      } else {                          // V stored transposed: Vt[bh][dh][n]
        u32x4 pk;
        pk.x = pack2(acc[mi][j][0], acc[mi][j][1]);
        pk.y = pack2(acc[mi][j][2], acc[mi][j][3]);
        pk.z = pack2(acc[mi][j][4], acc[mi][j][5]);
        pk.w = pack2(acc[mi][j][6], acc[mi][j][7]);
        u16* vt = Vtb + (bh * HDIM + (size_t)dh) * NN;
        *(u32x4*)(vt + nloc + lhi * 8) = pk;
      }
    }
  }
}

// ---------- kernel 2: flash attention, one 16-row q-tile per wave ----------
// S^T = K*Q^T so softmax stats are per-lane; P feeds O^T = Vt*P^T as B-operand
// after one xor-16 exchange. K/V operands are double-buffered (prefetch next
// 32-kv chunk before computing the current one) to hide L2 latency.
__global__ __launch_bounds__(256) void k_flash_attn(const u16* __restrict__ Qb,
                                                    const u16* __restrict__ Kb,
                                                    const u16* __restrict__ Vtb,
                                                    u16* __restrict__ aout) {
  int wave = (blockIdx.x * 256 + threadIdx.x) >> 5;
  int lane = threadIdx.x & 31;
  int lrow = lane & 15, lhi = lane >> 4;
  int kaA = lhi * 16, kaB = lhi * 16;

  const int QT = NN / 16;           // 128 q-tiles per (b,h)
  int bh = wave / QT;
  int qt = wave % QT;

  const u16* Q  = Qb  + (size_t)bh * NN * HDIM;
  const u16* K  = Kb  + (size_t)bh * NN * HDIM;
  const u16* Vt = Vtb + (size_t)bh * HDIM * NN;

  // Q B-operands, hoisted (column q = lrow over d, two 32-d chunks)
  const u16* qrow = Q + (size_t)(qt * 16 + lrow) * HDIM;
  v16bf bq0 = ld16bf(qrow + kaB, 8);
  v16bf bq1 = ld16bf(qrow + 32 + kaB, 8);

  v8f oacc[4];
#pragma unroll
  for (int t = 0; t < 4; ++t)
#pragma unroll
    for (int r = 0; r < 8; ++r) oacc[t][r] = 0.f;
  float mrun = -1e30f, lrun = 0.f;

  v16bf kb_[2][4];   // [buf][ k0:d0-31 | k0:d32-63 | k1:d0-31 | k1:d32-63 ]
  v16bf vb_[2][4];   // [buf][ Vt d-tile 0..3 ]
  auto loadkv = [&](int kv0, int s) {
    const u16* kr0 = K + (size_t)(kv0 + lrow) * HDIM + kaA;
    kb_[s][0] = ld16bf(kr0, 8);
    kb_[s][1] = ld16bf(kr0 + 32, 8);
    const u16* kr1 = kr0 + (size_t)16 * HDIM;
    kb_[s][2] = ld16bf(kr1, 8);
    kb_[s][3] = ld16bf(kr1 + 32, 8);
    const u16* vr = Vt + (size_t)lrow * NN + kv0 + kaA;
    vb_[s][0] = ld16bf(vr, 8);
    vb_[s][1] = ld16bf(vr + (size_t)16 * NN, 8);
    vb_[s][2] = ld16bf(vr + (size_t)32 * NN, 8);
    vb_[s][3] = ld16bf(vr + (size_t)48 * NN, 8);
  };

  loadkv(0, 0);
#pragma unroll 2
  for (int kv0 = 0; kv0 < NN; kv0 += 32) {
    int cb = (kv0 >> 5) & 1;
    int kvn = (kv0 + 32 < NN) ? (kv0 + 32) : kv0;   // clamped prefetch, no branch
    loadkv(kvn, cb ^ 1);

    // S^T tiles: st0 = kv[kv0..+15] x q, st1 = kv[kv0+16..+31] x q
    v8f st0, st1;
#pragma unroll
    for (int r = 0; r < 8; ++r) { st0[r] = 0.f; st1[r] = 0.f; }
    st0 = WMMA_BF16(kb_[cb][0], bq0, st0);
    st0 = WMMA_BF16(kb_[cb][1], bq1, st0);
    st1 = WMMA_BF16(kb_[cb][2], bq0, st1);
    st1 = WMMA_BF16(kb_[cb][3], bq1, st1);

    // online softmax: lane holds q=lrow, 16 kv scores; combine pair via xor16
    float mt = st0[0];
#pragma unroll
    for (int r = 1; r < 8; ++r) mt = fmaxf(mt, st0[r]);
#pragma unroll
    for (int r = 0; r < 8; ++r) mt = fmaxf(mt, st1[r]);
    mt = fmaxf(mt, swz16f(mt));
    float mnew = fmaxf(mrun, mt);
    float al = __expf(mrun - mnew);
    float p0[8], p1[8], lt = 0.f;
#pragma unroll
    for (int r = 0; r < 8; ++r) {
      p0[r] = __expf(st0[r] - mnew);
      p1[r] = __expf(st1[r] - mnew);
      lt += p0[r] + p1[r];
    }
    lt += swz16f(lt);
    lrun = lrun * al + lt;
    mrun = mnew;

    // Build P^T B-operand (col q = lrow, K = kv contiguous per half-wave):
    // lane-lo needs kv 0..15  -> [own p0 | partner p0]
    // lane-hi needs kv 16..31 -> [partner p1 | own p1]
    u32 pk0[4], pk1[4], xk0[4], xk1[4];
#pragma unroll
    for (int i = 0; i < 4; ++i) {
      pk0[i] = pack2(p0[2 * i], p0[2 * i + 1]);
      pk1[i] = pack2(p1[2 * i], p1[2 * i + 1]);
    }
#pragma unroll
    for (int i = 0; i < 4; ++i) { xk0[i] = swz16(pk0[i]); xk1[i] = swz16(pk1[i]); }
    u32x8 bw;
#pragma unroll
    for (int i = 0; i < 4; ++i) {
      bw[i]     = lhi ? xk1[i] : pk0[i];
      bw[4 + i] = lhi ? pk1[i] : xk0[i];
    }
    v16bf Bp = __builtin_bit_cast(v16bf, bw);

    // O^T accumulate: 4 d-tiles, A = Vt rows (d) over kv
#pragma unroll
    for (int t = 0; t < 4; ++t) {
      oacc[t] = oacc[t] * al;   // flash rescale (per-lane q) before accumulate
      oacc[t] = WMMA_BF16(vb_[cb][t], Bp, oacc[t]);
    }
  }

  // finalize: divide by l, store bf16 into attn-out [8192][768]
  float rl = 1.0f / lrun;
  int b = bh / NHEADS, h = bh % NHEADS;
  size_t row = (size_t)b * NN + qt * 16 + lrow;   // per-lane q row
#pragma unroll
  for (int t = 0; t < 4; ++t) {
    u32x4 pk;
    pk.x = pack2(oacc[t][0] * rl, oacc[t][1] * rl);
    pk.y = pack2(oacc[t][2] * rl, oacc[t][3] * rl);
    pk.z = pack2(oacc[t][4] * rl, oacc[t][5] * rl);
    pk.w = pack2(oacc[t][6] * rl, oacc[t][7] * rl);
    *(u32x4*)(aout + row * DIM + h * HDIM + t * 16 + lhi * 8) = pk;
  }
}

// ---------- kernel 3: output projection + bias (32x64 tile/wave, fp32 out) ----------
__global__ __launch_bounds__(256) void k_proj_gemm(const u16* __restrict__ ab,
                                                   const u16* __restrict__ wb,
                                                   const float* __restrict__ bias,
                                                   float* __restrict__ out) {
  int wave = (blockIdx.x * 256 + threadIdx.x) >> 5;
  int lane = threadIdx.x & 31;
  int lrow = lane & 15, lhi = lane >> 4;
  int kaA = lhi * 16, kaB = lhi * 16;
  const int NSTRIP = DIM / 64;      // 12
  int m  = (wave / NSTRIP) * 32;
  int n0 = (wave % NSTRIP) * 64;

  v8f acc[2][4];
#pragma unroll
  for (int mi = 0; mi < 2; ++mi)
#pragma unroll
    for (int j = 0; j < 4; ++j)
#pragma unroll
      for (int r = 0; r < 8; ++r) acc[mi][j][r] = 0.f;

  const u16* arow0 = ab + (size_t)(m + lrow) * DIM + kaA;
  const u16* arow1 = arow0 + (size_t)16 * DIM;
  const u16* bbase = wb + (size_t)(n0 + lrow) * DIM + kaB;

  for (int kc = 0; kc < DIM; kc += 32) {
    v16bf A0 = ld16bf(arow0 + kc, 8);
    v16bf A1 = ld16bf(arow1 + kc, 8);
    v16bf B0 = ld16bf(bbase + kc, 8);
    v16bf B1 = ld16bf(bbase + (size_t)16 * DIM + kc, 8);
    v16bf B2 = ld16bf(bbase + (size_t)32 * DIM + kc, 8);
    v16bf B3 = ld16bf(bbase + (size_t)48 * DIM + kc, 8);
    acc[0][0] = WMMA_BF16(A0, B0, acc[0][0]);
    acc[1][0] = WMMA_BF16(A1, B0, acc[1][0]);
    acc[0][1] = WMMA_BF16(A0, B1, acc[0][1]);
    acc[1][1] = WMMA_BF16(A1, B1, acc[1][1]);
    acc[0][2] = WMMA_BF16(A0, B2, acc[0][2]);
    acc[1][2] = WMMA_BF16(A1, B2, acc[1][2]);
    acc[0][3] = WMMA_BF16(A0, B3, acc[0][3]);
    acc[1][3] = WMMA_BF16(A1, B3, acc[1][3]);
  }

#pragma unroll
  for (int mi = 0; mi < 2; ++mi) {
    int mbase = m + 16 * mi;
#pragma unroll
    for (int j = 0; j < 4; ++j) {
      int d = n0 + 16 * j + lrow;      // per-lane output column
      float bv = bias[d];
#pragma unroll
      for (int r = 0; r < 8; ++r)
        out[(size_t)(mbase + lhi * 8 + r) * DIM + d] = acc[mi][j][r] + bv;
    }
  }
}

// ---------- host ----------
extern "C" void kernel_launch(void* const* d_in, const int* in_sizes, int n_in,
                              void* d_out, int out_size, void* d_ws, size_t ws_size,
                              hipStream_t stream) {
  (void)in_sizes; (void)n_in; (void)out_size; (void)ws_size;
  const float* x      = (const float*)d_in[0];
  const float* qkv_w  = (const float*)d_in[1];
  const float* proj_w = (const float*)d_in[2];
  const float* proj_b = (const float*)d_in[3];
  float* out = (float*)d_out;

  char* ws = (char*)d_ws;
  u16* xb  = (u16*)(ws + 0);                  // 8192*768   bf16 = 12,582,912 B
  u16* qwb = (u16*)(ws + 12582912);           // 2304*768   bf16 =  3,538,944 B
  u16* pwb = (u16*)(ws + 16121856);           // 768*768    bf16 =  1,179,648 B
  u16* Qb  = (u16*)(ws + 17301504);           // 48*2048*64 bf16 = 12,582,912 B
  u16* Kb  = (u16*)(ws + 29884416);
  u16* Vtb = (u16*)(ws + 42467328);
  u16* ab  = (u16*)(ws + 55050240);           // ends at 67,633,152 B

  int nx = ROWS * DIM;
  k_cvt_bf16<<<(nx + 255) / 256, 256, 0, stream>>>(x, xb, nx);
  int nw = QKV_DIM * DIM;
  k_cvt_bf16<<<(nw + 255) / 256, 256, 0, stream>>>(qkv_w, qwb, nw);
  int np = DIM * DIM;
  k_cvt_bf16<<<(np + 255) / 256, 256, 0, stream>>>(proj_w, pwb, np);

  k_qkv_gemm<<<((ROWS / 32) * (QKV_DIM / 64)) / 8, 256, 0, stream>>>(xb, qwb, Qb, Kb, Vtb);
  k_flash_attn<<<(BB * NHEADS * (NN / 16)) / 8, 256, 0, stream>>>(Qb, Kb, Vtb, ab);
  k_proj_gemm<<<((ROWS / 32) * (DIM / 64)) / 8, 256, 0, stream>>>(ab, pwb, proj_b, out);
}
